// MultiHeadAttention_51436528337630
// MI455X (gfx1250) — matrix operand order repaired and checked by reading, not compiled, not run on hardware
//
#include <hip/hip_runtime.h>

typedef __attribute__((ext_vector_type(16))) _Float16 v16h;
typedef __attribute__((ext_vector_type(8)))  _Float16 v8h;
typedef __attribute__((ext_vector_type(4)))  _Float16 v4h;
typedef __attribute__((ext_vector_type(8)))  float    v8f;
typedef __attribute__((ext_vector_type(4)))  int      v4i;

#define WMMA_F16(A, B, C) \
  __builtin_amdgcn_wmma_f32_16x16x32_f16(false, (A), false, (B), (short)0, (C), false, false)

constexpr int Bc  = 8;
constexpr int Sc  = 1024;
constexpr int Dc  = 768;
constexpr int Hc  = 12;
constexpr int DHc = 64;
constexpr int Mtot = Bc * Sc;   // 8192

// ---- CDNA5 async global->LDS staging (probe-confirmed builtin) ------------
#if __has_builtin(__builtin_amdgcn_global_load_async_to_lds_b128) && \
    __has_builtin(__builtin_amdgcn_s_wait_asynccnt)
#define HAVE_ASYNC_LDS 1
typedef __attribute__((address_space(1))) v4i gv4i;   // global int4
typedef __attribute__((address_space(3))) v4i sv4i;   // LDS int4
__device__ __forceinline__ void async_copy_b128(const _Float16* g, _Float16* l) {
  __builtin_amdgcn_global_load_async_to_lds_b128((gv4i*)g, (sv4i*)l, 0, 0);
}
#define ASYNC_WAIT0() __builtin_amdgcn_s_wait_asynccnt(0)
#else
#define HAVE_ASYNC_LDS 0
#define ASYNC_WAIT0()
#endif

// ---------------------------------------------------------------------------
// Weight f32 -> f16 conversion
// ---------------------------------------------------------------------------
__global__ void cvt_w_kernel(const float* __restrict__ src,
                             _Float16* __restrict__ dst, int n4) {
  int i = blockIdx.x * blockDim.x + threadIdx.x;
  if (i < n4) {
    float4 v = ((const float4*)src)[i];
    v4h h = { (_Float16)v.x, (_Float16)v.y, (_Float16)v.z, (_Float16)v.w };
    ((v4h*)dst)[i] = h;
  }
}

// ---------------------------------------------------------------------------
// WMMA GEMM: C[m][n] = sum_k A[m][k] * W[n][k] + bias[n]   (nn.Linear)
// Block tile 128x64, BK=32, double-buffered LDS, async staging for f16 tiles.
// 256 threads = 8 waves (4M x 2N), wave computes a 32x32 = 2x2 WMMA block.
// MODE 0: f16 out, (b,h,s,dh)   MODE 1: f16 out, (b,h,dh,s)   MODE 2: f32 out
// ---------------------------------------------------------------------------
template <int MODE, bool A16>
__global__ __launch_bounds__(256) void gemm_wmma_kernel(
    const float* __restrict__ Af32, const _Float16* __restrict__ Af16,
    const _Float16* __restrict__ W16, const float* __restrict__ bias,
    _Float16* __restrict__ out16, float* __restrict__ out32) {
  constexpr int STR = 40;                 // LDS row stride (halves)
  __shared__ _Float16 Xs[2][128 * STR];
  __shared__ _Float16 Wss[2][64 * STR];

  const int m0   = blockIdx.y * 128;
  const int n0   = blockIdx.x * 64;
  const int tid  = threadIdx.x;
  const int lane = tid & 31;
  const int wv   = tid >> 5;
  const int wm   = wv >> 1;
  const int wn   = wv & 1;
  const int lm   = lane & 15;
  const int hi   = lane >> 4;

  auto stage_x_load = [&](int kk, float4* xr) {
    #pragma unroll
    for (int it = 0; it < 4; ++it) {
      int q = tid + it * 256;
      int row = q >> 3, c4 = q & 7;
      xr[it] = *(const float4*)(Af32 + (size_t)(m0 + row) * Dc + kk + c4 * 4);
    }
  };
  auto stage_x_store = [&](_Float16* xs, const float4* xr) {
    #pragma unroll
    for (int it = 0; it < 4; ++it) {
      int q = tid + it * 256;
      int row = q >> 3, c4 = q & 7;
      v4h hv = { (_Float16)xr[it].x, (_Float16)xr[it].y,
                 (_Float16)xr[it].z, (_Float16)xr[it].w };
      *(v4h*)&xs[row * STR + c4 * 4] = hv;
    }
  };
  auto stage_x16 = [&](int kk, _Float16* xs) {
    #pragma unroll
    for (int it = 0; it < 2; ++it) {
      int q = tid + it * 256;
      int row = q >> 2, c8 = q & 3;
#if HAVE_ASYNC_LDS
      async_copy_b128(Af16 + (size_t)(m0 + row) * Dc + kk + c8 * 8,
                      xs + row * STR + c8 * 8);
#else
      v8h xv = *(const v8h*)(Af16 + (size_t)(m0 + row) * Dc + kk + c8 * 8);
      *(v8h*)(xs + row * STR + c8 * 8) = xv;
#endif
    }
  };
  auto stage_w = [&](int kk, _Float16* ws) {
    int row = tid >> 2, c8 = tid & 3;
#if HAVE_ASYNC_LDS
    async_copy_b128(W16 + (size_t)(n0 + row) * Dc + kk + c8 * 8,
                    ws + row * STR + c8 * 8);
#else
    v8h wv8 = *(const v8h*)(W16 + (size_t)(n0 + row) * Dc + kk + c8 * 8);
    *(v8h*)(ws + row * STR + c8 * 8) = wv8;
#endif
  };

  v8f acc[2][2] = {};

  // ---- prologue: stage first tile ----
  if constexpr (!A16) {
    float4 xr0[4];
    stage_x_load(0, xr0);
    stage_x_store(&Xs[0][0], xr0);
  } else {
    stage_x16(0, &Xs[0][0]);
  }
  stage_w(0, &Wss[0][0]);
  ASYNC_WAIT0();
  __syncthreads();

  for (int kk = 0; kk < Dc; kk += 32) {
    const int cur  = (kk >> 5) & 1;
    const int nxt  = cur ^ 1;
    const bool more = (kk + 32) < Dc;

    float4 xr[4];
    if (more) {
      if constexpr (!A16) stage_x_load(kk + 32, xr);
      else                stage_x16(kk + 32, &Xs[nxt][0]);
      stage_w(kk + 32, &Wss[nxt][0]);
    }

    // ---- fragments (gfx1250 16-bit A 16x32 / B 32x16 lane layouts) ----
    const _Float16* xcur = &Xs[cur][0];
    const _Float16* wcur = &Wss[cur][0];
    v16h a[2], bf[2];
    #pragma unroll
    for (int im = 0; im < 2; ++im) {
      const _Float16* p = xcur + (wm * 32 + im * 16 + lm) * STR + hi * 8;
      v8h lo = *(const v8h*)p;
      v8h hh = *(const v8h*)(p + 16);
      v16h t;
      #pragma unroll
      for (int i = 0; i < 8; ++i) { t[i] = lo[i]; t[i + 8] = hh[i]; }
      a[im] = t;
    }
    #pragma unroll
    for (int in = 0; in < 2; ++in) {
      const _Float16* p = wcur + (wn * 32 + in * 16 + lm) * STR + hi * 8;
      v8h b0 = *(const v8h*)p;
      v8h b1 = *(const v8h*)(p + 16);
      v16h t;
      #pragma unroll
      for (int i = 0; i < 8; ++i) { t[i] = b0[i]; t[i + 8] = b1[i]; }
      bf[in] = t;
    }
    #pragma unroll
    for (int im = 0; im < 2; ++im)
      #pragma unroll
      for (int in = 0; in < 2; ++in)
        acc[im][in] = WMMA_F16(a[im], bf[in], acc[im][in]);

    if (more) {
      if constexpr (!A16) stage_x_store(&Xs[nxt][0], xr);
    }
    ASYNC_WAIT0();
    __syncthreads();
  }

  #pragma unroll
  for (int im = 0; im < 2; ++im)
    #pragma unroll
    for (int in = 0; in < 2; ++in) {
      int gn = n0 + wn * 32 + in * 16 + lm;
      float bvs = bias[gn];
      #pragma unroll
      for (int r = 0; r < 8; ++r) {
        int gm = m0 + wm * 32 + im * 16 + hi * 8 + r;
        float val = acc[im][in][r] + bvs;
        if constexpr (MODE == 2) {
          out32[(size_t)gm * Dc + gn] = val;
        } else {
          int bb = gm >> 10, ss = gm & 1023;
          int hd = gn >> 6,  dh = gn & 63;
          if constexpr (MODE == 0)
            out16[(((size_t)(bb * Hc + hd)) * Sc + ss) * DHc + dh] = (_Float16)val;
          else
            out16[(((size_t)(bb * Hc + hd)) * DHc + dh) * Sc + ss] = (_Float16)val;
        }
      }
    }
}

// ---------------------------------------------------------------------------
// Flash attention, transposed formulation:
//   S^T = K @ Q^T  (lane = one query column -> scalar softmax state)
//   O^T = V^T @ P^T
// Grid (S/64, B*H), 128 threads = 4 waves, wave owns 16 queries.
// K/V 32-key tiles staged once per block in LDS (async, double-buffered);
// all 4 waves consume them -> 4x less L2 traffic than per-wave loads.
// Q,K: (b,h,s,dh) f16.  Vt: (b,h,dh,s) f16.  O: (b,s,d) f16.
// ---------------------------------------------------------------------------
__global__ __launch_bounds__(128) void attn_wmma_kernel(
    const _Float16* __restrict__ Q, const _Float16* __restrict__ Kt,
    const _Float16* __restrict__ Vt, _Float16* __restrict__ O) {
  constexpr float SCALE = 0.03608439182435161f;  // 1/sqrt(768) (module quirk)
  constexpr int KSTR = 72;                       // 32 keys x 64 dh, padded
  constexpr int VSTR = 40;                       // 64 dh x 32 keys, padded
  __shared__ _Float16 Ks[2][32 * KSTR];
  __shared__ _Float16 Vs[2][64 * VSTR];
  __shared__ _Float16 Ps[4][16 * 40];            // per-wave P^T bounce buffer

  const int bh = blockIdx.y;
  const int bb = bh / Hc;
  const int hd = bh % Hc;
  const int tid  = threadIdx.x;
  const int lane = tid & 31;
  const int wv   = tid >> 5;
  const int lm   = lane & 15;
  const int hi   = lane >> 4;

  const _Float16* Qb = Q  + (size_t)bh * (Sc * DHc);
  const _Float16* Kb = Kt + (size_t)bh * (Sc * DHc);
  const _Float16* Vb = Vt + (size_t)bh * (DHc * Sc);
  const int q0 = blockIdx.x * 64 + wv * 16;

  auto stage_kv = [&](int kt, _Float16* ks, _Float16* vs) {
    #pragma unroll
    for (int it = 0; it < 2; ++it) {
      int q = tid + it * 128;       // 256 b128 chunks each for K and V
      int kr = q >> 3, kc = q & 7;  // K: 32 rows(key) x 8 chunks(dh)
      int vr = q >> 2, vc = q & 3;  // V: 64 rows(dh)  x 4 chunks(key)
#if HAVE_ASYNC_LDS
      async_copy_b128(Kb + (size_t)(kt + kr) * DHc + kc * 8, ks + kr * KSTR + kc * 8);
      async_copy_b128(Vb + (size_t)vr * Sc + kt + vc * 8,    vs + vr * VSTR + vc * 8);
#else
      v8h kvv = *(const v8h*)(Kb + (size_t)(kt + kr) * DHc + kc * 8);
      *(v8h*)(ks + kr * KSTR + kc * 8) = kvv;
      v8h vvv = *(const v8h*)(Vb + (size_t)vr * Sc + kt + vc * 8);
      *(v8h*)(vs + vr * VSTR + vc * 8) = vvv;
#endif
    }
  };

  // B-fragments of Q^T (loop-invariant, SCALE pre-folded): k=dh, n=query
  v16h bq[2];
  #pragma unroll
  for (int ks = 0; ks < 2; ++ks) {
    const _Float16* p = Qb + (size_t)(q0 + lm) * DHc + ks * 32 + hi * 8;
    v8h b0 = *(const v8h*)p;
    v8h b1 = *(const v8h*)(p + 16);
    v16h t;
    #pragma unroll
    for (int i = 0; i < 8; ++i) {
      t[i]     = (_Float16)((float)b0[i] * SCALE);
      t[i + 8] = (_Float16)((float)b1[i] * SCALE);
    }
    bq[ks] = t;
  }

  float mrow = -1e30f, lrow = 0.0f;   // per-lane scalars (query = lm)
  v8f accO[4] = {};                   // O^T tiles: row dh = 16t + r + 8hi, col = lm
  _Float16* myP = &Ps[wv][0];

  // ---- prologue: stage first K/V tile ----
  stage_kv(0, &Ks[0][0], &Vs[0][0]);
  ASYNC_WAIT0();
  __syncthreads();

  for (int kt = 0; kt < Sc; kt += 32) {
    const int cur = (kt >> 5) & 1;
    if (kt + 32 < Sc) stage_kv(kt + 32, &Ks[cur ^ 1][0], &Vs[cur ^ 1][0]);
    const _Float16* kcur = &Ks[cur][0];
    const _Float16* vcur = &Vs[cur][0];

    // ---- S^T tiles: st[j] = K(16k x 64dh) @ Q^T(64dh x 16q) ----
    v8f st[2];
    #pragma unroll
    for (int j = 0; j < 2; ++j) {
      v8f c = {};
      #pragma unroll
      for (int ks = 0; ks < 2; ++ks) {
        const _Float16* p = kcur + (j * 16 + lm) * KSTR + ks * 32 + hi * 8;
        v8h lo = *(const v8h*)p;
        v8h hh = *(const v8h*)(p + 16);
        v16h a;
        #pragma unroll
        for (int i = 0; i < 8; ++i) { a[i] = lo[i]; a[i + 8] = hh[i]; }
        c = WMMA_F16(a, bq[ks], c);
      }
      st[j] = c;
    }

    // ---- online softmax: lane holds 16 scores of query lm ----
    float tmax = st[0][0];
    #pragma unroll
    for (int r = 1; r < 8; ++r) tmax = fmaxf(tmax, st[0][r]);
    #pragma unroll
    for (int r = 0; r < 8; ++r) tmax = fmaxf(tmax, st[1][r]);
    tmax = fmaxf(tmax, __shfl_xor(tmax, 16, 32));   // combine key halves

    float mn   = fmaxf(mrow, tmax);
    float corr = __expf(mrow - mn);
    mrow = mn;

    float rsum = 0.0f;
    #pragma unroll
    for (int j = 0; j < 2; ++j)
      #pragma unroll
      for (int r = 0; r < 8; ++r) {
        float pp = __expf(st[j][r] - mn);
        st[j][r] = pp;
        rsum += pp;
      }
    rsum += __shfl_xor(rsum, 16, 32);
    lrow = lrow * corr + rsum;
    #pragma unroll
    for (int t = 0; t < 4; ++t)
      #pragma unroll
      for (int r = 0; r < 8; ++r) accO[t][r] *= corr;

    // ---- pack P^T into LDS as Ps[query][key] (2x ds_store_b128) ----
    #pragma unroll
    for (int j = 0; j < 2; ++j) {
      v8h ph;
      #pragma unroll
      for (int r = 0; r < 8; ++r) ph[r] = (_Float16)st[j][r];
      *(v8h*)&myP[lm * 40 + j * 16 + hi * 8] = ph;
    }

    // ---- B-fragment of P^T (same-wave DS ops are in-order) ----
    v16h bp;
    {
      const _Float16* p = &myP[lm * 40 + hi * 8];
      v8h b0 = *(const v8h*)p;
      v8h b1 = *(const v8h*)(p + 16);
      #pragma unroll
      for (int i = 0; i < 8; ++i) { bp[i] = b0[i]; bp[i + 8] = b1[i]; }
    }

    // ---- O^T += V^T(16dh x 32k) @ P^T(32k x 16q), per dh tile ----
    #pragma unroll
    for (int t = 0; t < 4; ++t) {
      const _Float16* p = vcur + (t * 16 + lm) * VSTR + hi * 8;
      v8h lo = *(const v8h*)p;
      v8h hh = *(const v8h*)(p + 16);
      v16h a;
      #pragma unroll
      for (int i = 0; i < 8; ++i) { a[i] = lo[i]; a[i + 8] = hh[i]; }
      accO[t] = WMMA_F16(a, bp, accO[t]);
    }

    ASYNC_WAIT0();
    __syncthreads();
  }

  // ---- normalize, pack, store (4x global_store_b128 per lane) ----
  float inv = 1.0f / lrow;
  #pragma unroll
  for (int t = 0; t < 4; ++t) {
    v8h oh;
    #pragma unroll
    for (int r = 0; r < 8; ++r) oh[r] = (_Float16)(accO[t][r] * inv);
    *(v8h*)(O + ((size_t)(bb * Sc + q0 + lm)) * Dc + hd * DHc + t * 16 + hi * 8) = oh;
  }
}

// ---------------------------------------------------------------------------
extern "C" void kernel_launch(void* const* d_in, const int* in_sizes, int n_in,
                              void* d_out, int out_size, void* d_ws, size_t ws_size,
                              hipStream_t stream) {
  (void)in_sizes; (void)n_in; (void)out_size; (void)ws_size;

  const float* queries = (const float*)d_in[0];
  const float* keys    = (const float*)d_in[1];
  const float* values  = (const float*)d_in[2];
  const float* Wq = (const float*)d_in[3];
  const float* bq = (const float*)d_in[4];
  const float* Wk = (const float*)d_in[5];
  const float* bk = (const float*)d_in[6];
  const float* Wv = (const float*)d_in[7];
  const float* bv = (const float*)d_in[8];
  const float* Wo = (const float*)d_in[9];
  const float* bo = (const float*)d_in[10];
  float* out = (float*)d_out;

  const size_t nW = (size_t)Dc * Dc;              // 589824
  const size_t nT = (size_t)Bc * Hc * Sc * DHc;   // 6291456
  _Float16* Wq16 = (_Float16*)d_ws;
  _Float16* Wk16 = Wq16 + nW;
  _Float16* Wv16 = Wk16 + nW;
  _Float16* Wo16 = Wv16 + nW;
  _Float16* Q16  = Wo16 + nW;
  _Float16* K16  = Q16 + nT;
  _Float16* Vt16 = K16 + nT;
  _Float16* O16  = Vt16 + nT;

  {
    int n4 = (int)(nW / 4);
    int blocks = (n4 + 255) / 256;
    cvt_w_kernel<<<blocks, 256, 0, stream>>>(Wq, Wq16, n4);
    cvt_w_kernel<<<blocks, 256, 0, stream>>>(Wk, Wk16, n4);
    cvt_w_kernel<<<blocks, 256, 0, stream>>>(Wv, Wv16, n4);
    cvt_w_kernel<<<blocks, 256, 0, stream>>>(Wo, Wo16, n4);
  }

  dim3 ggrid(Dc / 64, Mtot / 128);  // (12, 64)
  gemm_wmma_kernel<0, false><<<ggrid, 256, 0, stream>>>(queries, nullptr, Wq16, bq, Q16, nullptr);
  gemm_wmma_kernel<0, false><<<ggrid, 256, 0, stream>>>(keys,    nullptr, Wk16, bk, K16, nullptr);
  gemm_wmma_kernel<1, false><<<ggrid, 256, 0, stream>>>(values,  nullptr, Wv16, bv, Vt16, nullptr);

  dim3 agrid(Sc / 64, Bc * Hc);     // (16, 96)
  attn_wmma_kernel<<<agrid, 128, 0, stream>>>(Q16, K16, Vt16, O16);

  gemm_wmma_kernel<2, true><<<ggrid, 256, 0, stream>>>(nullptr, O16, Wo16, bo, nullptr, out);
}
